// RSNN_33981781246501
// MI455X (gfx1250) — matrix operand order repaired and checked by reading, not compiled, not run on hardware
//
#include <hip/hip_runtime.h>
#include <hip/hip_bf16.h>

// Model constants
#define B_    32
#define T_    256
#define NIN_  1024
#define N_    640          // NCELL + NHID
#define D_    16
#define THR_  0.4f
#define TEMP_ 2.0f

#define NTILES   40        // N_/16
#define NBLK     80        // 2 mtiles * 40 ntiles, persistent grid
#define KB_REC   320       // (N_*D_)/32 k-blocks per N-tile
#define KB_IN    32        // NIN_/32 k-blocks
#define RWAVES   8         // K-split waves per block in recurrent kernel

typedef __attribute__((ext_vector_type(16))) __bf16 v16bf;
typedef __attribute__((ext_vector_type(8)))  float  v8f;

union Frag { v16bf v; uint4 q[2]; unsigned short u[16]; };

__device__ __forceinline__ unsigned short f2bf(float f) {
  unsigned x = __float_as_uint(f);
  x += 0x7FFFu + ((x >> 16) & 1u);           // round-to-nearest-even
  return (unsigned short)(x >> 16);
}

// CDNA5 async global->LDS copy of one 32B lane slice (two b128 DMAs, ASYNCcnt).
// ISA 15.18: the immediate offset is added to both the LDS and the global address,
// so a single offset:16 covers both sides of the second half.
__device__ __forceinline__ void async_copy32(unsigned ldsaddr, const unsigned short* g) {
  unsigned long long ga = (unsigned long long)(size_t)g;
  asm volatile("global_load_async_to_lds_b128 %0, %1, off"
               :: "v"(ldsaddr), "v"(ga) : "memory");
  asm volatile("global_load_async_to_lds_b128 %0, %1, off offset:16"
               :: "v"(ldsaddr), "v"(ga) : "memory");
}

#if __has_builtin(__builtin_amdgcn_s_wait_asynccnt)
#define WAIT_ASYNC(n) __builtin_amdgcn_s_wait_asynccnt(n)
#else
#define WAIT_ASYNC(n) asm volatile("s_wait_asynccnt %0" :: "i"(n) : "memory")
#endif

// ---------------------------------------------------------------- prologue
__global__ void zero_cnt(unsigned* c) { if (threadIdx.x == 0) *c = 0u; }

// Swizzle w_in [NIN][N] f32 -> bf16 B-fragments: [ntile][kb][lane][e]
__global__ void swz_win(const float* __restrict__ w_in, unsigned short* __restrict__ out) {
  int idx = blockIdx.x * 256 + threadIdx.x;                 // 40*32*32*16 = 655360
  if (idx >= NTILES * KB_IN * 32 * 16) return;
  int e     = idx & 15;
  int lane  = (idx >> 4) & 31;
  int kb    = (idx >> 9) & 31;
  int ntile = idx >> 14;
  int n = lane & 15;
  int k = kb * 32 + ((lane < 16) ? 0 : 8) + ((e < 8) ? e : e + 8); // B[k][n]: K=2r,2r+1 per VGPR
  out[idx] = f2bf(w_in[(size_t)k * N_ + ntile * 16 + n]);
}

// Swizzle w_rec [j][i][d] f32 -> bf16 B-fragments over k=d*640+i: [ntile][kb][lane][e]
__global__ void swz_wrec(const float* __restrict__ w_rec, unsigned short* __restrict__ out) {
  int idx = blockIdx.x * 256 + threadIdx.x;                 // 40*320*32*16 = 6553600
  if (idx >= NTILES * KB_REC * 32 * 16) return;
  int e     = idx & 15;
  int lane  = (idx >> 4) & 31;
  int kb    = (idx >> 9) % KB_REC;
  int ntile = idx / (KB_REC * 512);
  int n = lane & 15;
  int k = kb * 32 + ((lane < 16) ? 0 : 8) + ((e < 8) ? e : e + 8); // k in [0, 10240)
  int d = k / N_, i = k % N_;
  int j = ntile * 16 + n;
  out[idx] = f2bf(w_rec[((size_t)j * N_ + i) * D_ + d]);
}

// ------------------------------------------------- input projection GEMM
// Ipre[t][b][j] = x[b][t][:] @ w_in[:,j] + bias[j]   (M=8192, N=640, K=1024)
__global__ __launch_bounds__(128) void in_gemm(const float* __restrict__ x,
                                               const unsigned short* __restrict__ winswz,
                                               const float* __restrict__ bias,
                                               float* __restrict__ ipre) {
  int wave = threadIdx.x >> 5, lane = threadIdx.x & 31;
  int tile = blockIdx.x * 4 + wave;                         // [0, 20480)
  int mtile = tile / NTILES, ntile = tile % NTILES;
  int m = lane & 15;
  int koff = (lane < 16) ? 0 : 8;                           // 16-bit A layout
  const float* xr = x + (size_t)(mtile * 16 + m) * NIN_;
  v8f c = {};
  for (int kb = 0; kb < KB_IN; ++kb) {
    Frag a, b;
    const float* p = xr + kb * 32 + koff;
    float4 f0 = *(const float4*)(p);
    float4 f1 = *(const float4*)(p + 4);
    float4 f2 = *(const float4*)(p + 16);
    float4 f3 = *(const float4*)(p + 20);
    a.u[0]=f2bf(f0.x);  a.u[1]=f2bf(f0.y);  a.u[2]=f2bf(f0.z);  a.u[3]=f2bf(f0.w);
    a.u[4]=f2bf(f1.x);  a.u[5]=f2bf(f1.y);  a.u[6]=f2bf(f1.z);  a.u[7]=f2bf(f1.w);
    a.u[8]=f2bf(f2.x);  a.u[9]=f2bf(f2.y);  a.u[10]=f2bf(f2.z); a.u[11]=f2bf(f2.w);
    a.u[12]=f2bf(f3.x); a.u[13]=f2bf(f3.y); a.u[14]=f2bf(f3.z); a.u[15]=f2bf(f3.w);
    const uint4* bp = (const uint4*)(winswz + (((size_t)ntile * KB_IN + kb) * 32 + lane) * 16);
    b.q[0] = bp[0]; b.q[1] = bp[1];
    c = __builtin_amdgcn_wmma_f32_16x16x32_bf16(false, a.v, false, b.v, (short)0, c, false, false);
  }
  int n = lane & 15;
  int mofs = (lane < 16) ? 0 : 8;
  int j = ntile * 16 + n;
  float bj = bias[j];
#pragma unroll
  for (int r = 0; r < 8; ++r) {
    int row = mtile * 16 + r + mofs;                        // row = b*T + t (x layout)
    int bb = row >> 8, t = row & 255;
    ipre[((size_t)t * B_ + bb) * N_ + j] = c[r] + bj;
  }
}

// ------------------------------------------------- persistent recurrent loop
__global__ __launch_bounds__(256) void rsnn_steps(const unsigned short* __restrict__ wrswz,
                                                  const float* __restrict__ ipre,
                                                  const float* __restrict__ u,
                                                  float* __restrict__ out,
                                                  unsigned short* __restrict__ zhist,
                                                  unsigned* __restrict__ cnt) {
  __shared__ float red[RWAVES * 32 * 8];                    // 8 KB cross-wave partials
  __shared__ uint4 bstage[RWAVES * 2 * 64];                 // 16 KB: per-wave double buffer
  const int wave = threadIdx.x >> 5, lane = threadIdx.x & 31;
  const int mtile = blockIdx.x / NTILES;                    // 0..1
  const int ntile = blockIdx.x % NTILES;                    // 0..39
  const int m = lane & 15;
  const int koff = (lane < 16) ? 0 : 8;
  const size_t BTN = (size_t)B_ * T_ * N_;

  uint4* myst = &bstage[wave * 128];                        // 2 x 1KB buffers
  unsigned lds_lo = (unsigned)(size_t)(&myst[lane * 2]);       // buf0 lane slot
  unsigned lds_hi = (unsigned)(size_t)(&myst[64 + lane * 2]);  // buf1 lane slot

  for (int t = 0; t < T_; ++t) {
    v8f c = {};
    int d0 = (t >= D_) ? 0 : (D_ - t);                      // history exists for d >= D-t
    for (int d = d0 + wave; d < D_; d += RWAVES) {          // K split across 8 waves
      const unsigned short* zrow =
          zhist + ((size_t)(t - D_ + d) * B_ + (mtile * 16 + m)) * N_;
      const unsigned short* wp =
          wrswz + ((size_t)ntile * KB_REC + d * 20) * 512 + (size_t)lane * 16;
      async_copy32(lds_lo, wp);                             // prologue: stage cb=0
      int buf = 0;
      for (int cb = 0; cb < 20; ++cb) {                     // 20 k-blocks per delay slot
        if (cb + 1 < 20) {                                  // stage next while computing
          async_copy32(buf ? lds_lo : lds_hi, wp + (size_t)(cb + 1) * 512);
          WAIT_ASYNC(2);                                    // current stage landed
        } else {
          WAIT_ASYNC(0);
        }
        Frag a, b;
        a.q[0] = *(const uint4*)(zrow + cb * 32 + koff);
        a.q[1] = *(const uint4*)(zrow + cb * 32 + koff + 16);
        const uint4* s = buf ? &myst[64 + lane * 2] : &myst[lane * 2];
        b.q[0] = s[0]; b.q[1] = s[1];
        c = __builtin_amdgcn_wmma_f32_16x16x32_bf16(false, a.v, false, b.v, (short)0, c, false, false);
        buf ^= 1;
      }
    }
#pragma unroll
    for (int r = 0; r < 8; ++r) red[(wave * 32 + lane) * 8 + r] = c[r];
    __syncthreads();
    if (threadIdx.x < 32) {                                 // wave 0: reduce + spike epilogue
      int n = lane & 15;
      int mofs = (lane < 16) ? 0 : 8;
      int j = ntile * 16 + n;
#pragma unroll
      for (int r = 0; r < 8; ++r) {
        float irec = 0.0f;
#pragma unroll
        for (int w = 0; w < RWAVES; ++w) irec += red[(w * 32 + lane) * 8 + r];
        int bb = mtile * 16 + r + mofs;
        float v  = irec + ipre[((size_t)t * B_ + bb) * N_ + j];
        float vs = (v - THR_) * (1.0f / THR_);
        float sig = 1.0f / (1.0f + __expf(-TEMP_ * vs));
        size_t o = ((size_t)bb * T_ + t) * N_ + j;
        float z = (sig > u[o]) ? 1.0f : 0.0f;
        out[o]           = z;
        out[BTN + o]     = v;
        out[2 * BTN + o] = vs;
        zhist[((size_t)t * B_ + bb) * N_ + j] = (z != 0.0f) ? (unsigned short)0x3F80u : (unsigned short)0u;
      }
    }
    // device-wide barrier between time steps (80 blocks, all resident)
    __threadfence();
    if (threadIdx.x == 0) {
      __hip_atomic_fetch_add(cnt, 1u, __ATOMIC_ACQ_REL, __HIP_MEMORY_SCOPE_AGENT);
      unsigned target = (unsigned)NBLK * (unsigned)(t + 1);
      while (__hip_atomic_load(cnt, __ATOMIC_ACQUIRE, __HIP_MEMORY_SCOPE_AGENT) < target)
        __builtin_amdgcn_s_sleep(2);
    }
    __syncthreads();
    __threadfence();                                        // acquire side for all threads
  }
}

// ---------------------------------------------------------------- launch
extern "C" void kernel_launch(void* const* d_in, const int* in_sizes, int n_in,
                              void* d_out, int out_size, void* d_ws, size_t ws_size,
                              hipStream_t stream) {
  const float* x     = (const float*)d_in[0];   // [32,256,1024]
  const float* u     = (const float*)d_in[1];   // [32,256,640]
  const float* w_in  = (const float*)d_in[2];   // [1024,640]
  const float* w_rec = (const float*)d_in[3];   // [640,640,16]
  const float* bias  = (const float*)d_in[4];   // [640]
  float* out = (float*)d_out;

  char* ws = (char*)d_ws;
  size_t off = 0;
  unsigned* cnt = (unsigned*)(ws + off);                 off += 256;
  unsigned short* winswz = (unsigned short*)(ws + off);  off += (size_t)NTILES * KB_IN  * 512 * 2; // 1.31 MB
  unsigned short* wrswz  = (unsigned short*)(ws + off);  off += (size_t)NTILES * KB_REC * 512 * 2; // 13.1 MB
  float* ipre = (float*)(ws + off);                      off += (size_t)T_ * B_ * N_ * 4;          // 21.0 MB
  unsigned short* zhist = (unsigned short*)(ws + off);   off += (size_t)T_ * B_ * N_ * 2;          // 10.5 MB

  zero_cnt<<<1, 32, 0, stream>>>(cnt);
  swz_win <<<(NTILES * KB_IN  * 512 + 255) / 256, 256, 0, stream>>>(w_in, winswz);
  swz_wrec<<<(NTILES * KB_REC * 512 + 255) / 256, 256, 0, stream>>>(w_rec, wrswz);
  in_gemm <<<(512 * NTILES) / 4, 128, 0, stream>>>(x, winswz, bias, ipre);
  rsnn_steps<<<NBLK, 256, 0, stream>>>(wrswz, ipre, u, out, zhist, cnt);
}
